// ConvBlock_27908697489432
// MI455X (gfx1250) — matrix operand order repaired and checked by reading, not compiled, not run on hardware
//
#include <hip/hip_runtime.h>
#include <hip/hip_bf16.h>
#include <math.h>

typedef __bf16 bf16_t;
typedef __attribute__((ext_vector_type(16))) __bf16 v16bf;
typedef __attribute__((ext_vector_type(8)))  float  v8f;
typedef __attribute__((ext_vector_type(4)))  int    v4i;

#define B_      2
#define N_      1024
#define H_      64
#define D_      32
#define K_      32768          /* N_ * D_ */
#define MT_     64             /* M-tile rows per workgroup */
#define KSPLIT_ 16
#define KCHUNK_ (K_ / KSPLIT_) /* 2048 */
#define KSTEP_  64             /* K advanced per barrier interval */
#define VT_BYTES ((size_t)B_ * H_ * K_ * 2)   /* 8 MiB bf16 */

// ---------------------------------------------------------------------------
// Kernel 0: vT[b][c][k] = bf16( x[b, s, c] * W[d, c] ),  k = s*32 + d
// Column-major (c-major, k contiguous) == exact WMMA B-fragment feed layout.
// ---------------------------------------------------------------------------
__global__ __launch_bounds__(256) void prep_vT(const float* __restrict__ x,
                                               const float* __restrict__ W,
                                               bf16_t* __restrict__ vT) {
  size_t idx = (size_t)blockIdx.x * 256 + threadIdx.x;   // < 2^22
  int k = (int)(idx & (K_ - 1));
  int c = (int)((idx >> 15) & (H_ - 1));
  int b = (int)(idx >> 21);
  int s = k >> 5;
  int d = k & 31;
  float val = x[((size_t)b * N_ + s) * H_ + c] * W[d * H_ + c];
  vT[idx] = (bf16_t)val;
}

// ---------------------------------------------------------------------------
// Kernel 1: split-K GEMM  partial[b][ks][r][c] = KB[b,r,kchunk] @ vT[b][kchunk,c]
// 256 threads = 8 waves. Wave (wm = w&1, wn = w>>1) owns a 32-row x 16-col
// output strip: 2 A-fragments (LDS) x 1 B-fragment (global, L2-resident),
// minimizing L2 B traffic. 64x64 bf16 A tile double-buffered in LDS,
// one barrier per 64-K step.
// ---------------------------------------------------------------------------
__global__ __launch_bounds__(256) void gemm_split(const float* __restrict__ KB,
                                                  const bf16_t* __restrict__ vT,
                                                  float* __restrict__ partial) {
  __shared__ __align__(16) bf16_t As[2][MT_ * KSTEP_];   // 2 x 8 KiB

  const int ks = blockIdx.x, mtile = blockIdx.y, b = blockIdx.z;
  const int t = threadIdx.x;
  const int lane = t & 31, wave = t >> 5;
  const int wm = wave & 1;        // 32-row group
  const int wn = wave >> 1;       // 16-col group
  const int kbase = ks * KCHUNK_;

  // cooperative A staging: thread t loads row t/4, 16 floats at col (t%4)*16
  const int arow = t >> 2;
  const int acol = (t & 3) * 16;
  const float* aptr =
      KB + ((size_t)(b * N_ + mtile * MT_ + arow)) * K_ + kbase + acol;
  const int s_byte = arow * (KSTEP_ * 2) + acol * 2;

  // A-fragment LDS addressing (ISA 16-bit A 16x32 layout), row stride 128 B:
  //   lane<16 : bytes [ko*2+0,+16) = K0..7 , [+32,+48) = K16..23  of row lane&15
  //   lane>=16: bytes [ko*2+16,+32) = K8..15, [+48,+64) = K24..31
  const int fb0 = (wm * 32 + (lane & 15)) * (KSTEP_ * 2) + ((lane >> 4) << 4);
  const int fb1 = fb0 + 16 * (KSTEP_ * 2);

  // B-fragment: lane holds K 8h..8h+7 and 16+8h..16+8h+7 of column c; vT is [c][k] contiguous
  const bf16_t* bp = vT + ((size_t)(b * H_ + wn * 16 + (lane & 15))) * K_ +
                     kbase + ((lane >> 4) << 3);

  v8f acc0 = {};   // rows wm*32 + 0..15
  v8f acc1 = {};   // rows wm*32 + 16..31

  float4 f0 = ((const float4*)(aptr +  0))[0];
  float4 f1 = ((const float4*)(aptr +  4))[0];
  float4 f2 = ((const float4*)(aptr +  8))[0];
  float4 f3 = ((const float4*)(aptr + 12))[0];

  for (int kk = 0; kk < KCHUNK_; kk += KSTEP_) {
    const int buf = (kk >> 6) & 1;
    char* sb = (char*)&As[buf][0];

    union { bf16_t h[8]; v4i q; } pk0, pk1;
    pk0.h[0] = (bf16_t)f0.x; pk0.h[1] = (bf16_t)f0.y;
    pk0.h[2] = (bf16_t)f0.z; pk0.h[3] = (bf16_t)f0.w;
    pk0.h[4] = (bf16_t)f1.x; pk0.h[5] = (bf16_t)f1.y;
    pk0.h[6] = (bf16_t)f1.z; pk0.h[7] = (bf16_t)f1.w;
    pk1.h[0] = (bf16_t)f2.x; pk1.h[1] = (bf16_t)f2.y;
    pk1.h[2] = (bf16_t)f2.z; pk1.h[3] = (bf16_t)f2.w;
    pk1.h[4] = (bf16_t)f3.x; pk1.h[5] = (bf16_t)f3.y;
    pk1.h[6] = (bf16_t)f3.z; pk1.h[7] = (bf16_t)f3.w;
    *(v4i*)(sb + s_byte)      = pk0.q;
    *(v4i*)(sb + s_byte + 16) = pk1.q;

    if (kk + KSTEP_ < KCHUNK_) {          // prefetch next A slab (uniform)
      f0 = ((const float4*)(aptr + kk + KSTEP_ +  0))[0];
      f1 = ((const float4*)(aptr + kk + KSTEP_ +  4))[0];
      f2 = ((const float4*)(aptr + kk + KSTEP_ +  8))[0];
      f3 = ((const float4*)(aptr + kk + KSTEP_ + 12))[0];
    }
    __syncthreads();

    const char* rb = (const char*)&As[buf][0];
#pragma unroll
    for (int ko = 0; ko < KSTEP_; ko += 32) {
      union { v4i q[2]; v16bf v; } a0, a1, bf;
      a0.q[0] = *(const v4i*)(rb + fb0 + ko * 2);
      a0.q[1] = *(const v4i*)(rb + fb0 + ko * 2 + 32);
      a1.q[0] = *(const v4i*)(rb + fb1 + ko * 2);
      a1.q[1] = *(const v4i*)(rb + fb1 + ko * 2 + 32);
      bf.q[0] = *(const v4i*)(bp + kk + ko);
      bf.q[1] = *(const v4i*)(bp + kk + ko + 16);

      acc0 = __builtin_amdgcn_wmma_f32_16x16x32_bf16(
          false, a0.v, false, bf.v, (short)0, acc0, false, false);
      acc1 = __builtin_amdgcn_wmma_f32_16x16x32_bf16(
          false, a1.v, false, bf.v, (short)0, acc1, false, false);
    }
  }

  // C/D layout: VGPR v -> row M = v + (lane>=16 ? 8 : 0), col N = lane&15
  float* pout =
      partial + (((size_t)(b * KSPLIT_ + ks)) * N_ + mtile * MT_) * H_;
  const int rbase = wm * 32 + ((lane >> 4) << 3);
  const int cc = wn * 16 + (lane & 15);
#pragma unroll
  for (int v = 0; v < 8; ++v) {
    pout[(rbase + v) * H_ + cc]        = acc0[v];
    pout[(rbase + 16 + v) * H_ + cc]   = acc1[v];
  }
}

// ---------------------------------------------------------------------------
// Kernel 2: reduce split-K partials + conv_bias -> LayerNorm -> MLP(GELU)
// One block per 32 rows. All intermediates in LDS (padded strides).
// ---------------------------------------------------------------------------
__device__ __forceinline__ float gelu_tanh(float v) {
  float v3 = v * v * v;
  return 0.5f * v * (1.0f + tanhf(0.7978845608028654f * (v + 0.044715f * v3)));
}

__global__ __launch_bounds__(256) void fuse_ln_mlp(
    const float* __restrict__ partial, const float* __restrict__ conv_bias,
    const float* __restrict__ ln_scale, const float* __restrict__ ln_bias,
    const float* __restrict__ W1, const float* __restrict__ b1,
    const float* __restrict__ W2, const float* __restrict__ b2,
    float* __restrict__ out) {
  const int ROWS = 32;
  __shared__ float a_s[32][H_ + 1];     // stride 65: conflict-free row scans
  __shared__ float h_s[32][264];        // stride 264: rows 8 banks apart
  __shared__ float mu_s[32], rs_s[32];

  const int t = threadIdx.x;
  const int b = blockIdx.x >> 5;
  const int r0 = (blockIdx.x & 31) * ROWS;

  // reduce split-K partials + bias
#pragma unroll
  for (int i = 0; i < 8; ++i) {
    int idx = i * 256 + t;
    int r = idx >> 6, c = idx & 63;
    float acc = conv_bias[c];
    const float* p =
        partial + ((size_t)b * KSPLIT_ * N_ + (r0 + r)) * H_ + c;
#pragma unroll
    for (int ksp = 0; ksp < KSPLIT_; ++ksp)
      acc += p[(size_t)ksp * N_ * H_];
    a_s[r][c] = acc;
  }
  __syncthreads();

  // per-row LayerNorm stats
  if (t < ROWS) {
    float m = 0.f;
#pragma unroll
    for (int c = 0; c < H_; ++c) m += a_s[t][c];
    m *= (1.0f / H_);
    float v = 0.f;
#pragma unroll
    for (int c = 0; c < H_; ++c) { float d = a_s[t][c] - m; v += d * d; }
    v *= (1.0f / H_);
    mu_s[t] = m;
    rs_s[t] = rsqrtf(v + 1e-6f);
  }
  __syncthreads();

#pragma unroll
  for (int i = 0; i < 8; ++i) {
    int idx = i * 256 + t;
    int r = idx >> 6, c = idx & 63;
    a_s[r][c] = (a_s[r][c] - mu_s[r]) * rs_s[r] * ln_scale[c] + ln_bias[c];
  }
  __syncthreads();

  // GEMM1 (64 -> 256) + GELU: one hidden unit per thread
  {
    const int j = t;
    float w1c[H_];
#pragma unroll
    for (int c = 0; c < H_; ++c) w1c[c] = W1[c * 256 + j];
    const float bj = b1[j];
    for (int r = 0; r < ROWS; ++r) {
      float acc = bj;
#pragma unroll
      for (int c = 0; c < H_; ++c) acc += a_s[r][c] * w1c[c];
      h_s[r][j] = gelu_tanh(acc);
    }
  }
  __syncthreads();

  // GEMM2 (256 -> 64): thread -> (row t>>3, 8 output cols)
  {
    const int r = t >> 3;
    const int cg = (t & 7) * 8;
    float acc[8];
#pragma unroll
    for (int i = 0; i < 8; ++i) acc[i] = b2[cg + i];
    for (int j = 0; j < 256; ++j) {
      float hv = h_s[r][j];
#pragma unroll
      for (int i = 0; i < 8; ++i) acc[i] += hv * W2[j * 64 + cg + i];
    }
    float* op = out + ((size_t)b * N_ + r0 + r) * H_ + cg;
#pragma unroll
    for (int i = 0; i < 8; ++i) op[i] = acc[i];
  }
}

// ---------------------------------------------------------------------------
extern "C" void kernel_launch(void* const* d_in, const int* in_sizes, int n_in,
                              void* d_out, int out_size, void* d_ws,
                              size_t ws_size, hipStream_t stream) {
  const float* x         = (const float*)d_in[0];
  const float* KB        = (const float*)d_in[1];
  const float* W         = (const float*)d_in[2];
  const float* conv_bias = (const float*)d_in[3];
  const float* ln_scale  = (const float*)d_in[4];
  const float* ln_bias   = (const float*)d_in[5];
  const float* W1        = (const float*)d_in[6];
  const float* b1        = (const float*)d_in[7];
  const float* W2        = (const float*)d_in[8];
  const float* b2        = (const float*)d_in[9];
  float* out = (float*)d_out;

  bf16_t* vT     = (bf16_t*)d_ws;                          // 8 MiB
  float* partial = (float*)((char*)d_ws + VT_BYTES);       // 8 MiB

  prep_vT<<<(B_ * H_ * K_) / 256, 256, 0, stream>>>(x, W, vT);
  gemm_split<<<dim3(KSPLIT_, N_ / MT_, B_), 256, 0, stream>>>(KB, vT, partial);
  fuse_ln_mlp<<<B_ * (N_ / 32), 256, 0, stream>>>(
      partial, conv_bias, ln_scale, ln_bias, W1, b1, W2, b2, out);
}
